// CorrelationLayer_90151363543756
// MI455X (gfx1250) — matrix operand order repaired and checked by reading, not compiled, not run on hardware
//
#include <hip/hip_runtime.h>
#include <hip/hip_bf16.h>
#include <stddef.h>

// FlowNet-C correlation on MI455X (gfx1250), bf16 WMMA band-GEMM formulation.
//
// corr(b,dx,dy,h,w) = sum_c x1[b,c,h,w] * x2[b,c,h+dy,w+dx]  (zero padded),
// then 2x2 avg pool and /C.  For fixed (b,h,dy) this is a banded W x W GEMM
// with K=C=128 on v_wmma_f32_16x16x32_bf16, 16x16 tiles covering only the
// |dx|<=20 band (4 N-tiles per M-tile).  All global accesses use a uniform
// SGPR base + unsigned 32-bit lane offset so the backend emits saddr-form
// global_load_b128 (no per-lane 64-bit address chains).

typedef __attribute__((ext_vector_type(16))) __bf16 v16bf;
typedef __attribute__((ext_vector_type(8)))  __bf16 v8bf;
typedef __attribute__((ext_vector_type(8)))  float  v8f;

#define BATCH 4
#define CHN   128
#define HH    96
#define WW    96
#define MD    20
#define KD    41          // 2*MD+1
#define OHH   48
#define OWW   48
#define PH    (HH + 2*MD) // 136 padded rows
#define PW    144         // 20 left pad + 96 + 28 right pad (N-tiles reach padded col w0+63)
#define PADL  20
#define MTILES 6          // 96/16

static constexpr size_t X1_ELEMS  = (size_t)BATCH * HH * WW * CHN;      // 4,718,592
static constexpr size_t X2P_ELEMS = (size_t)BATCH * PH * PW * CHN;      // 10,027,008

// ---- pre-pass: BCHW f32 -> BHWC bf16 (x1), and zero-padded BHWC bf16 (x2) ----

__global__ void cvt_x1_kernel(const float* __restrict__ src,
                              __hip_bfloat16* __restrict__ dst) {
  size_t idx = (size_t)blockIdx.x * blockDim.x + threadIdx.x;
  if (idx >= X1_ELEMS) return;
  int c = (int)(idx & (CHN - 1));
  size_t t = idx >> 7;
  int w = (int)(t % WW); t /= WW;
  int h = (int)(t % HH);
  int b = (int)(t / HH);
  dst[idx] = __float2bfloat16(src[(((size_t)b * CHN + c) * HH + h) * WW + w]);
}

__global__ void cvt_x2_kernel(const float* __restrict__ src,
                              __hip_bfloat16* __restrict__ dst) {
  size_t idx = (size_t)blockIdx.x * blockDim.x + threadIdx.x;
  if (idx >= X2P_ELEMS) return;
  int c = (int)(idx & (CHN - 1));
  size_t t = idx >> 7;
  int w = (int)(t % PW); t /= PW;
  int h = (int)(t % PH);
  int b = (int)(t / PH);
  float v = 0.0f;
  int hs = h - MD;
  int ws = w - PADL;
  if (hs >= 0 && hs < HH && ws >= 0 && ws < WW)
    v = src[(((size_t)b * CHN + c) * HH + hs) * WW + ws];
  dst[idx] = __float2bfloat16(v);
}

// ---- main kernel: one workgroup per (b, oh, M-tile); 8 waves split the 41 dy ----

__global__ __launch_bounds__(256) void corr_wmma_kernel(
    const __hip_bfloat16* __restrict__ x1bf,
    const __hip_bfloat16* __restrict__ x2bf,
    float* __restrict__ out) {
  // per-wave scratch for D tiles: 16 rows (w) x 64 cols (w' band) f32
  __shared__ float lds[8][16][64];

  const int lane = threadIdx.x & 31;
  const int wv   = threadIdx.x >> 5;
  const int mt   = blockIdx.x % MTILES;
  const int oh   = (blockIdx.x / MTILES) % OHH;
  const int b    = blockIdx.x / (MTILES * OHH);
  const int w0   = mt * 16;
  const int lm   = lane & 15;   // matrix row/col within tile
  const int lh   = lane >> 4;   // lane half selects K sub-range

  // A tiles: x1 rows h = 2*oh, 2*oh+1; A is 16(M=w) x 32(K=c) per chunk.
  // ISA layout: lanes 0-15 carry K = kc*32 + {0..7, 16..23}; lanes 16-31 the
  // other K octets.  Channels-last makes each half one contiguous 16B load.
  v16bf amat[2][4];
#pragma unroll
  for (int h2 = 0; h2 < 2; ++h2) {
    const unsigned arow =
        (unsigned)(((b * HH + (2 * oh + h2)) * WW + (w0 + lm)) * CHN);
#pragma unroll
    for (int kc = 0; kc < 4; ++kc) {
      const unsigned c0 = arow + (unsigned)(kc * 32 + lh * 8);
      v8bf lo = *(const v8bf*)(x1bf + c0);
      v8bf hi = *(const v8bf*)(x1bf + c0 + 16u);
      amat[h2][kc] = __builtin_shufflevector(
          lo, hi, 0, 1, 2, 3, 4, 5, 6, 7, 8, 9, 10, 11, 12, 13, 14, 15);
    }
  }

  for (int j = wv; j < KD; j += 8) {  // j = dy index, wave-uniform
    // prefetch next dy row of x2 for this wave (global_prefetch_b8)
    if (j + 8 < KD)
      __builtin_prefetch(
          (const void*)(x2bf +
                        (unsigned)(((b * PH + (2 * oh + j + 8)) * PW + w0) * CHN)),
          0, 1);

#pragma unroll
    for (int nt = 0; nt < 4; ++nt) {
      v8f acc = {};  // accumulates BOTH h rows -> pooling over h for free
#pragma unroll
      for (int h2 = 0; h2 < 2; ++h2) {
        // padded x2 row = (2*oh + h2) + j ; N columns start at w0 - 20
        const unsigned brow =
            (unsigned)(((b * PH + (2 * oh + h2 + j)) * PW +
                        (w0 + nt * 16 + lm)) * CHN);
        // Batch all 4 K-chunk loads of this B row before the 4 WMMAs so the
        // scheduler can clause them and use descending loadcnt waits.
        v16bf bmat[4];
#pragma unroll
        for (int kc = 0; kc < 4; ++kc) {
          // B 32x16: lanes 0-15 hold K=kc*32+{0..7,16..23}, lanes 16-31 K=kc*32+{8..15,24..31}
          const unsigned c0 = brow + (unsigned)(kc * 32 + lh * 8);
          v8bf lo = *(const v8bf*)(x2bf + c0);
          v8bf hi = *(const v8bf*)(x2bf + c0 + 16u);
          bmat[kc] = __builtin_shufflevector(
              lo, hi, 0, 1, 2, 3, 4, 5, 6, 7, 8, 9, 10, 11, 12, 13, 14, 15);
        }
#pragma unroll
        for (int kc = 0; kc < 4; ++kc) {
          acc = __builtin_amdgcn_wmma_f32_16x16x32_bf16(
              false, amat[h2][kc], false, bmat[kc], (short)0, acc, false, false);
        }
      }
      // D layout: VGPR r, lanes 0-15 -> (M=r, N=lane); lanes 16-31 -> (M=r+8)
#pragma unroll
      for (int vr = 0; vr < 8; ++vr)
        lds[wv][lh * 8 + vr][nt * 16 + lm] = acc[vr];
    }

    // Pool over w (diagonal pairs) and store.  Same-wave LDS ops are in-order,
    // so no barrier needed; EXEC divergence here reconverges before next WMMA.
#pragma unroll 1
    for (int t = 0; t < 11; ++t) {
      int idx = (t << 5) + lane;          // covers 41*8 = 328 outputs
      if (idx < KD * 8) {
        int di = idx >> 3;                // dx index 0..40  (dx = di - 20)
        int ol = idx & 7;                 // local ow
        int m0 = ol * 2;                  // band col q = (dx+20) + m = di + m
        float s = lds[wv][m0][di + m0] + lds[wv][m0 + 1][di + m0 + 1];
        out[(((size_t)b * (KD * KD) + di * KD + j) * OHH + oh) * OWW +
            (mt * 8 + ol)] = s * (1.0f / 512.0f);  // /4 (pool) /128 (C)
      }
    }
  }
}

extern "C" void kernel_launch(void* const* d_in, const int* in_sizes, int n_in,
                              void* d_out, int out_size, void* d_ws, size_t ws_size,
                              hipStream_t stream) {
  (void)in_sizes; (void)n_in; (void)out_size; (void)ws_size;
  const float* x1 = (const float*)d_in[0];
  const float* x2 = (const float*)d_in[1];
  float* out = (float*)d_out;

  char* ws = (char*)d_ws;  // needs ~28.2 MiB
  __hip_bfloat16* x1bf = (__hip_bfloat16*)ws;
  __hip_bfloat16* x2bf = (__hip_bfloat16*)(ws + X1_ELEMS * sizeof(__hip_bfloat16));

  cvt_x1_kernel<<<(unsigned)((X1_ELEMS + 255) / 256), 256, 0, stream>>>(x1, x1bf);
  cvt_x2_kernel<<<(unsigned)((X2P_ELEMS + 255) / 256), 256, 0, stream>>>(x2, x2bf);
  corr_wmma_kernel<<<BATCH * OHH * MTILES, 256, 0, stream>>>(x1bf, x2bf, out);
}
